// CBoEModule_43069932045059
// MI455X (gfx1250) — matrix operand order repaired and checked by reading, not compiled, not run on hardware
//
#include <hip/hip_runtime.h>
#include <hip/hip_bf16.h>

typedef __attribute__((ext_vector_type(16))) __bf16 v16bf;
typedef __attribute__((ext_vector_type(8)))  float  v8f;
typedef unsigned short u16;
typedef unsigned int   u32;

#define T_DIM  4096
#define N_DIM  32768
#define D_DIM  1024
#define MBLK   32      // query rows per workgroup (2 m-groups of 16)
#define KBLK   32      // keys per iteration
#define DSLICE 256     // D columns per wave (4 d-waves)
#define KTP    40      // padded K^T row stride in ushorts (80B, 16B-aligned)

#if __has_builtin(__builtin_amdgcn_sched_barrier)
#define SCHED_FENCE() __builtin_amdgcn_sched_barrier(0)
#else
#define SCHED_FENCE()
#endif

struct U32x8 { uint4 lo, hi; };

static __device__ __forceinline__ v16bf ld_frag(const u16* p0, const u16* p1) {
  U32x8 u;
  u.lo = *(const uint4*)p0;   // 16 B  -> ds_load_b128
  u.hi = *(const uint4*)p1;   // 16 B  -> ds_load_b128
  return __builtin_bit_cast(v16bf, u);
}

static __device__ __forceinline__ u16 f2bf(float f) {
  return __builtin_bit_cast(u16, (__bf16)f);
}

__global__ __launch_bounds__(256) void
flash_attn_bf16_wmma(const float* __restrict__ X,
                     const float* __restrict__ E,
                     float* __restrict__ O)
{
  __shared__ u16   sh_k[KBLK * D_DIM];   // 64 KB : K block, row-major bf16
  __shared__ u16   sh_u[D_DIM * KTP];    // 80 KB : Q tile at init, K^T in loop
  __shared__ float sh_sc[2][16][32];     //  4 KB : reduced score tiles

  const int t     = threadIdx.x;
  const int lane  = t & 31;
  const int wid   = t >> 5;
  const int mg    = wid >> 2;        // 0..1  : which 16-row query tile
  const int wd    = wid & 3;         // 0..3  : D-slice index
  const int dbase = wd * DSLICE;
  const int qblk  = blockIdx.x;
  const int rin   = lane & 15;       // row/col-in-tile
  const int half  = lane >> 4;       // lane half

  // ---- Q tile (32 x 1024 fp32) -> LDS bf16 ----
  {
    const float4* qg = (const float4*)(X + (size_t)qblk * MBLK * D_DIM);
    #pragma unroll 4
    for (int i = 0; i < 32; ++i) {
      float4 v = qg[i * 256 + t];
      u32 lo = (u32)f2bf(v.x) | ((u32)f2bf(v.y) << 16);
      u32 hi = (u32)f2bf(v.z) | ((u32)f2bf(v.w) << 16);
      *(uint2*)&sh_u[i * D_DIM + t * 4] = make_uint2(lo, hi);
    }
  }
  __syncthreads();

  // ---- hoist persistent Q fragments (A-matrix 16x32 layout) ----
  v16bf qf[8];
  {
    const int qrow = mg * 16 + rin;
    #pragma unroll
    for (int c = 0; c < 8; ++c) {
      const int d0 = dbase + c * 32 + half * 8;
      qf[c] = ld_frag(&sh_u[qrow * D_DIM + d0], &sh_u[qrow * D_DIM + d0 + 16]);
    }
  }

  v8f acc[16];
  #pragma unroll
  for (int i = 0; i < 16; ++i)
    #pragma unroll
    for (int r = 0; r < 8; ++r) acc[i][r] = 0.f;

  float m_run = -INFINITY;
  float l_run = 0.f;

  for (int kb = 0; kb < N_DIM / KBLK; ++kb) {
    __syncthreads();   // previous iteration's LDS consumers are done

    // zero score-reduction buffer (1024 floats / 256 threads)
    ((float*)sh_sc)[t]       = 0.f;
    ((float*)sh_sc)[t + 256] = 0.f;
    ((float*)sh_sc)[t + 512] = 0.f;
    ((float*)sh_sc)[t + 768] = 0.f;

    const float* eg = E + (size_t)kb * KBLK * D_DIM;
    if (kb + 1 < N_DIM / KBLK)
      __builtin_prefetch(eg + KBLK * D_DIM + t * 128, 0, 0);  // whole next 128KB block

    // ---- cooperative load: K block fp32 -> bf16, row-major + padded transpose ----
    #pragma unroll 4
    for (int i = 0; i < 32; ++i) {
      const int u   = i * 256 + t;
      const int key = (u >> 9) * 2;   // key pair
      const int d   = (u & 511) * 2;  // d pair
      float2 a = *(const float2*)(eg + key * D_DIM + d);
      float2 b = *(const float2*)(eg + (key + 1) * D_DIM + d);
      u16 a0 = f2bf(a.x), a1 = f2bf(a.y);
      u16 b0 = f2bf(b.x), b1 = f2bf(b.y);
      *(u32*)&sh_k[key * D_DIM + d]       = (u32)a0 | ((u32)a1 << 16);
      *(u32*)&sh_k[(key + 1) * D_DIM + d] = (u32)b0 | ((u32)b1 << 16);
      *(u32*)&sh_u[d * KTP + key]         = (u32)a0 | ((u32)b0 << 16);
      *(u32*)&sh_u[(d + 1) * KTP + key]   = (u32)a1 | ((u32)b1 << 16);
    }
    __syncthreads();

    // ---- partial scores: Q(16 x 256-slice) x K^T -> 16x32, split-D reduce in LDS ----
    #pragma unroll
    for (int nt = 0; nt < 2; ++nt) {
      const int keyrow = nt * 16 + rin;
      // Batch all 16 ds_load_b128 before the WMMAs (sched_barrier keeps them
      // hoisted -> 8 distinct live fragments, incremental dscnt waits).
      v16bf kfa[8];
      #pragma unroll
      for (int c = 0; c < 8; ++c) {
        const int d0 = dbase + c * 32 + half * 8;
        kfa[c] = ld_frag(&sh_k[keyrow * D_DIM + d0], &sh_k[keyrow * D_DIM + d0 + 16]);
      }
      SCHED_FENCE();
      // Two independent 4-deep accumulation chains hide the XDL
      // accumulation latency of a single serial 8-deep chain.
      v8f s0, s1;
      #pragma unroll
      for (int r = 0; r < 8; ++r) { s0[r] = 0.f; s1[r] = 0.f; }
      #pragma unroll
      for (int c = 0; c < 8; c += 2) {
        s0 = __builtin_amdgcn_wmma_f32_16x16x32_bf16(false, qf[c],     false, kfa[c],
                                                     (short)0, s0, false, false);
        s1 = __builtin_amdgcn_wmma_f32_16x16x32_bf16(false, qf[c + 1], false, kfa[c + 1],
                                                     (short)0, s1, false, false);
      }
      SCHED_FENCE();
      #pragma unroll
      for (int r = 0; r < 8; ++r)
        atomicAdd(&sh_sc[mg][r + half * 8][keyrow], s0[r] + s1[r]);   // ds_add_f32
    }
    __syncthreads();

    // ---- online softmax for row `rin` (lane pair L / L+16 covers all 32 keys) ----
    float sv[16];
    const float* srow = &sh_sc[mg][rin][0];
    #pragma unroll
    for (int v = 0; v < 8; ++v) {
      const int k0 = (v < 4) ? (2 * v + 8 * half) : (16 + 2 * (v - 4) + 8 * half);
      float2 p = *(const float2*)(srow + k0);
      sv[2 * v]     = p.x;
      sv[2 * v + 1] = p.y;
    }
    float lmax = sv[0];
    #pragma unroll
    for (int i = 1; i < 16; ++i) lmax = fmaxf(lmax, sv[i]);
    lmax = fmaxf(lmax, __shfl_xor(lmax, 16));
    const float newm  = fmaxf(m_run, lmax);
    const float alpha = __expf(m_run - newm);
    float psum = 0.f;
    v16bf pf;   // P fragment already in A-matrix 16x32 layout
    #pragma unroll
    for (int i = 0; i < 16; ++i) {
      float p = __expf(sv[i] - newm);
      psum += p;
      pf[i] = (__bf16)p;
    }
    psum += __shfl_xor(psum, 16);
    l_run = l_run * alpha + psum;
    m_run = newm;

    // rescale accumulators (rows r + 8*half live in this lane's acc VGPRs)
    float asc[8];
    #pragma unroll
    for (int r = 0; r < 8; ++r) asc[r] = __shfl(alpha, r + 8 * half);
    #pragma unroll
    for (int dt = 0; dt < 16; ++dt)
      #pragma unroll
      for (int r = 0; r < 8; ++r) acc[dt][r] *= asc[r];

    // ---- PV: P(16x32) x V(32 x 16-per-tile), 2 groups of 8: 16 batched LDS
    //      loads, then 8 INDEPENDENT WMMAs (distinct accumulators) ----
    #pragma unroll
    for (int g = 0; g < 2; ++g) {
      v16bf vfa[8];
      #pragma unroll
      for (int j = 0; j < 8; ++j) {
        const int dt = g * 8 + j;
        const int dd = dbase + dt * 16 + rin;           // lane = N (d column)
        const u16* bp = &sh_u[dd * KTP + half * 8];     // keys contiguous in K^T row
        vfa[j] = ld_frag(bp, bp + 16);
      }
      SCHED_FENCE();
      #pragma unroll
      for (int j = 0; j < 8; ++j) {
        const int dt = g * 8 + j;
        acc[dt] = __builtin_amdgcn_wmma_f32_16x16x32_bf16(false, pf, false, vfa[j],
                                                          (short)0, acc[dt], false, false);
      }
      SCHED_FENCE();
    }
  }

  // ---- epilogue: divide by softmax denominator, store fp32 ----
  const float inv = 1.f / l_run;
  float rsc[8];
  #pragma unroll
  for (int r = 0; r < 8; ++r) rsc[r] = __shfl(inv, r + 8 * half);
  float* og = O + ((size_t)qblk * MBLK + mg * 16) * D_DIM;
  #pragma unroll
  for (int dt = 0; dt < 16; ++dt) {
    const int dcol = dbase + dt * 16 + rin;
    #pragma unroll
    for (int r = 0; r < 8; ++r)
      og[(size_t)(r + half * 8) * D_DIM + dcol] = acc[dt][r] * rsc[r];
  }
}

extern "C" void kernel_launch(void* const* d_in, const int* in_sizes, int n_in,
                              void* d_out, int out_size, void* d_ws, size_t ws_size,
                              hipStream_t stream) {
  (void)in_sizes; (void)n_in; (void)d_ws; (void)ws_size; (void)out_size;
  const float* x = (const float*)d_in[0];
  const float* e = (const float*)d_in[1];
  float* o = (float*)d_out;
  dim3 grid(T_DIM / MBLK);
  dim3 block(256);
  hipLaunchKernelGGL(flash_attn_bf16_wmma, grid, block, 0, stream, x, e, o);
}
